// graphnetwork_54838142435720
// MI455X (gfx1250) — matrix operand order repaired and checked by reading, not compiled, not run on hardware
//
#include <hip/hip_runtime.h>
#include <hip/hip_bf16.h>
#include <stdint.h>

// ---------- types ----------
typedef __attribute__((ext_vector_type(16))) __bf16 v16bf;
typedef __attribute__((ext_vector_type(8)))  float  v8f;

union FragU {
    v16bf v;
    uint4 u[2];
};

static __device__ __forceinline__ unsigned short f2bf_bits(float f) {
    unsigned u = __float_as_uint(f);
    unsigned r = u + 0x7FFFu + ((u >> 16) & 1u);   // RNE
    return (unsigned short)(r >> 16);
}

// ---------- degree ----------
__global__ void deg_kernel(const int* __restrict__ dst, float* __restrict__ deg, int E) {
    int i = blockIdx.x * blockDim.x + threadIdx.x;
    if (i >= E) return;
    __hip_atomic_fetch_add(&deg[dst[i]], 1.0f, __ATOMIC_RELAXED, __HIP_MEMORY_SCOPE_AGENT);
}

__global__ void deginv_kernel(const float* __restrict__ deg, float* __restrict__ dinv, int N) {
    int i = blockIdx.x * blockDim.x + threadIdx.x;
    if (i >= N) return;
    dinv[i] = 1.0f / fmaxf(deg[i], 1.0f);
}

// ---------- edge scatter-add: agg[dst] += h[src] ----------
__global__ void scatter_kernel(const float* __restrict__ h, const int* __restrict__ src,
                               const int* __restrict__ dst, float* __restrict__ agg,
                               int E, int fin) {
    int idx = blockIdx.x * blockDim.x + threadIdx.x;
    int cpe = fin >> 2;            // float4 chunks per edge
    int tot = E * cpe;
    if (idx >= tot) return;
    int e = idx / cpe;
    int c = (idx - e * cpe) << 2;
    int s = src[e], d = dst[e];
    float4 v = *(const float4*)(h + (size_t)s * fin + c);
    float* p = agg + (size_t)d * fin + c;
    __hip_atomic_fetch_add(p + 0, v.x, __ATOMIC_RELAXED, __HIP_MEMORY_SCOPE_AGENT);
    __hip_atomic_fetch_add(p + 1, v.y, __ATOMIC_RELAXED, __HIP_MEMORY_SCOPE_AGENT);
    __hip_atomic_fetch_add(p + 2, v.z, __ATOMIC_RELAXED, __HIP_MEMORY_SCOPE_AGENT);
    __hip_atomic_fetch_add(p + 3, v.w, __ATOMIC_RELAXED, __HIP_MEMORY_SCOPE_AGENT);
}

// ---------- combined weight -> bf16, n-major: WcT[n*K + k], K = 2*fin ----------
__global__ void convw_kernel(const float* __restrict__ Ws, const float* __restrict__ Wn,
                             unsigned short* __restrict__ WcT, int fin, int fout) {
    int K = 2 * fin;
    int tot = K * fout;
    int idx = blockIdx.x * blockDim.x + threadIdx.x;
    if (idx >= tot) return;
    int n = idx / K;
    int k = idx - n * K;
    float v = (k < fin) ? Ws[(size_t)k * fout + n] : Wn[(size_t)(k - fin) * fout + n];
    WcT[idx] = f2bf_bits(v);
}

// ---------- concat tail copy: hnext[:, fout:fout+fin] = hin ----------
__global__ void concat_copy(const float* __restrict__ hin, float* __restrict__ hnext,
                            int N, int fin, int fout, int wout) {
    int idx = blockIdx.x * blockDim.x + threadIdx.x;
    int tot = N * fin;
    if (idx >= tot) return;
    int i = idx / fin;
    int k = idx - i * fin;
    hnext[(size_t)i * wout + fout + k] = hin[(size_t)i * fin + k];
}

// ---------- fused SAGE GEMM: out = [h | agg*dinv] @ Wc + b  (bf16 WMMA, f32 acc) ----------
// K = 2*FIN; A fragments built from f32 global loads converted to bf16 in regs.
template <int FIN, int NOUT, bool RELU>
__global__ __launch_bounds__(256) void sage_gemm(const float* __restrict__ h,
                                                 const float* __restrict__ agg,
                                                 const float* __restrict__ dinv,
                                                 const unsigned short* __restrict__ WcT,
                                                 const float* __restrict__ bias,
                                                 float* __restrict__ out, int ldOut,
                                                 int nMTiles) {
    constexpr int K  = 2 * FIN;
    constexpr int NT = NOUT / 16;
    __shared__ unsigned short sW[K * NOUT];

    // cooperative LDS fill of combined weights (bf16, n-major)
    const int tid = threadIdx.x;
    for (int i = tid; i < (K * NOUT) / 8; i += 256)
        ((uint4*)sW)[i] = ((const uint4*)WcT)[i];
    __syncthreads();

    const int wid   = tid >> 5;
    const int lane  = tid & 31;
    const int gtile = blockIdx.x * 8 + wid;
    const int total = nMTiles * NT;
    if (gtile >= total) return;                 // wave-uniform: EXEC stays all-ones

    const int mTile = gtile / NT;
    const int nTile = gtile - mTile * NT;

    const int row   = mTile * 16 + (lane & 15); // A-matrix row for this lane
    const int abase = (lane < 16) ? 0 : 8;      // ISA 16-bit A layout K sub-base
    const int n     = nTile * 16 + (lane & 15); // B/C/D column for this lane
    const int kb    = (lane < 16) ? 0 : 8;      // B layout: lanes 16-31 hold K+8..15, K+24..31

    const float dscale = dinv[row];
    const unsigned short* wrow = sW + n * K;

    v8f acc = {};
#pragma unroll
    for (int kk = 0; kk < K; kk += 32) {
        // ---- A fragment: f32 -> bf16 in registers ----
        const bool  inAgg = (kk >= FIN);        // FIN is a multiple of 32: chunk is pure
        const float* ap   = inAgg ? (agg + (size_t)row * FIN + (kk - FIN))
                                  : (h   + (size_t)row * FIN + kk);
        const float scale = inAgg ? dscale : 1.0f;
        float4 f0 = *(const float4*)(ap + abase);
        float4 f1 = *(const float4*)(ap + abase + 4);
        float4 f2 = *(const float4*)(ap + 16 + abase);
        float4 f3 = *(const float4*)(ap + 16 + abase + 4);
        FragU a;
        a.v[0]  = (__bf16)(f0.x * scale); a.v[1]  = (__bf16)(f0.y * scale);
        a.v[2]  = (__bf16)(f0.z * scale); a.v[3]  = (__bf16)(f0.w * scale);
        a.v[4]  = (__bf16)(f1.x * scale); a.v[5]  = (__bf16)(f1.y * scale);
        a.v[6]  = (__bf16)(f1.z * scale); a.v[7]  = (__bf16)(f1.w * scale);
        a.v[8]  = (__bf16)(f2.x * scale); a.v[9]  = (__bf16)(f2.y * scale);
        a.v[10] = (__bf16)(f2.z * scale); a.v[11] = (__bf16)(f2.w * scale);
        a.v[12] = (__bf16)(f3.x * scale); a.v[13] = (__bf16)(f3.y * scale);
        a.v[14] = (__bf16)(f3.z * scale); a.v[15] = (__bf16)(f3.w * scale);

        // ---- B fragment: two contiguous ds_load_b128 from n-major LDS weights ----
        FragU b;
        b.u[0] = *(const uint4*)(wrow + kk + kb);
        b.u[1] = *(const uint4*)(wrow + kk + kb + 16);

        acc = __builtin_amdgcn_wmma_f32_16x16x32_bf16(false, a.v, false, b.v,
                                                      (short)0, acc, false, false);
    }

    // ---- epilogue: C/D layout -> strided global store with bias (+ReLU) ----
    const float bn = bias[n];
    const int   r0 = mTile * 16 + ((lane < 16) ? 0 : 8);
#pragma unroll
    for (int r = 0; r < 8; ++r) {
        float v = acc[r] + bn;
        if (RELU) v = fmaxf(v, 0.0f);
        out[(size_t)(r0 + r) * ldOut + n] = v;
    }
}

// ---------- host ----------
static inline unsigned gdiv(long long n) { return (unsigned)((n + 255) / 256); }

extern "C" void kernel_launch(void* const* d_in, const int* in_sizes, int n_in,
                              void* d_out, int out_size, void* d_ws, size_t ws_size,
                              hipStream_t stream) {
    const int N = 100000;
    const int E = in_sizes[1];

    const float* in_feat = (const float*)d_in[0];
    const int*   src     = (const int*)d_in[1];
    const int*   dst     = (const int*)d_in[2];
    const float *Ws[6], *Wn[6], *bi[6];
    for (int i = 0; i < 6; ++i) {
        Ws[i] = (const float*)d_in[3 + 3 * i];
        Wn[i] = (const float*)d_in[4 + 3 * i];
        bi[i] = (const float*)d_in[5 + 3 * i];
    }

    char* ws = (char*)d_ws;
    const size_t MB = 1024ull * 1024ull;
    float*          deg  = (float*)(ws + 0);
    float*          dinv = (float*)(ws + 512 * 1024);
    float*          hA   = (float*)(ws + 1 * MB);
    float*          hB   = (float*)(ws + 53 * MB);
    float*          agg  = (float*)(ws + 105 * MB);
    unsigned short* WcT  = (unsigned short*)(ws + 157 * MB);

    // degrees
    hipMemsetAsync(deg, 0, (size_t)N * 4, stream);
    deg_kernel<<<gdiv(E), 256, 0, stream>>>(dst, deg, E);
    deginv_kernel<<<gdiv(N), 256, 0, stream>>>(deg, dinv, N);

    const int nMTiles = N / 16;

#define RUN_LAYER(L, FIN, NOUT, RELU, CONCAT, HIN, HOUT, WOUT)                               \
    do {                                                                                     \
        convw_kernel<<<gdiv(2 * FIN * NOUT), 256, 0, stream>>>(Ws[L], Wn[L], WcT, FIN, NOUT);\
        hipMemsetAsync(agg, 0, (size_t)N * FIN * 4, stream);                                 \
        scatter_kernel<<<gdiv((long long)E * (FIN / 4)), 256, 0, stream>>>(                  \
            HIN, src, dst, agg, E, FIN);                                                     \
        sage_gemm<FIN, NOUT, RELU><<<gdiv((long long)nMTiles * (NOUT / 16) * 32), 256, 0,    \
                                     stream>>>(HIN, agg, dinv, WcT, bi[L], HOUT, WOUT,       \
                                               nMTiles);                                     \
        if (CONCAT)                                                                          \
            concat_copy<<<gdiv((long long)N * FIN), 256, 0, stream>>>(HIN, HOUT, N, FIN,     \
                                                                      NOUT, WOUT);           \
    } while (0)

    float* outF = (float*)d_out;
    // L1: 64 -> 128, relu
    RUN_LAYER(0, 64, 128, true, false, in_feat, hA, 128);
    // L2: 128 -> 64, relu
    RUN_LAYER(1, 128, 64, true, false, hA, hB, 64);
    // L3: 64 -> 32, relu
    RUN_LAYER(2, 64, 32, true, false, hB, hA, 32);
    // L4: 32 -> 32, relu, concat h3 -> width 64
    RUN_LAYER(3, 32, 32, true, true, hA, hB, 64);
    // L5: 64 -> 64, relu, concat h4 -> width 128
    RUN_LAYER(4, 64, 64, true, true, hB, hA, 128);
    // L6: 128 -> 64, no relu, direct to output
    RUN_LAYER(5, 128, 64, false, false, hA, outF, 64);

#undef RUN_LAYER
}
